// MSRGNN_35759897707101
// MI455X (gfx1250) — matrix operand order repaired and checked by reading, not compiled, not run on hardware
//
#include <hip/hip_runtime.h>
#include <math.h>

// ---------------------------------------------------------------------------
// Types for CDNA5 WMMA (wave32): v_wmma_f32_16x16x32_bf16
// ---------------------------------------------------------------------------
typedef __attribute__((ext_vector_type(16))) __bf16        bf16x16;
typedef __attribute__((ext_vector_type(8)))  float         f32x8;
typedef __attribute__((ext_vector_type(4)))  float         f32x4;
typedef __attribute__((ext_vector_type(4)))  unsigned int  u32x4;

union BFVec {
  unsigned short us[16];
  bf16x16        v;
  u32x4          q[2];
};
static_assert(sizeof(BFVec) == 32, "BFVec size");

__device__ inline unsigned short bfbits(float f) {
  unsigned int u = __builtin_bit_cast(unsigned int, f);
  // round-to-nearest-even fp32 -> bf16
  return (unsigned short)((u + 0x7FFFu + ((u >> 16) & 1u)) >> 16);
}

__device__ inline float gelu_exact(float x) {
  return 0.5f * x * (1.0f + erff(x * 0.70710678118654752f));
}

enum { MODE_DIRECT = 0, MODE_EDGE_PW = 1, MODE_EDGE_AT = 2, MODE_UP = 3 };
enum { EPI_NONE = 0, EPI_GELU = 1, EPI_LEAKY = 2, EPI_ADD = 3 };

struct GemmArgs {
  const float* A;            // direct-mode A, row stride == K
  const unsigned short* WP;  // bf16 weights packed in B-fragment order
  const float* bias;         // [Ncols]
  float*       C;            // out [M, Ncols]
  int M, K, Ncols;
  int epi;
  const int* src;            // [E]
  const int* dst;            // [E]
  const float* IX;           // integrated features [N, D]
  int D, F, P;
  const float* AGG;          // aggregation buffer [N, Hagg]
  int Hagg;
};

// Resolve the base pointer of an 8-element contiguous k-run of gathered row r.
// All concat boundaries (F, 2F, 2F+P, D) are multiples of 8, so a run starting
// at k % 8 == 0 never crosses a region boundary; all bases are 16B aligned.
template <int MODE>
__device__ inline const float* rowptr8(const GemmArgs& g, int r, int k) {
  if (MODE == MODE_DIRECT) {
    return g.A + (size_t)r * g.K + k;
  } else if (MODE == MODE_EDGE_PW) {     // cat(ix[dst], ix[src])
    int h2 = (k >= g.D);
    int node = h2 ? g.src[r] : g.dst[r];
    int kk = h2 ? (k - g.D) : k;
    return g.IX + (size_t)node * g.D + kk;
  } else if (MODE == MODE_EDGE_AT) {     // cat(x_i, x_j, pos_i, pos_j)
    int F = g.F, P = g.P, node, kk;
    if (k < F)               { node = g.dst[r]; kk = k; }
    else if (k < 2 * F)      { node = g.src[r]; kk = k - F; }
    else if (k < 2 * F + P)  { node = g.dst[r]; kk = F + (k - 2 * F); }
    else                     { node = g.src[r]; kk = F + (k - 2 * F - P); }
    return g.IX + (size_t)node * (F + P) + kk;
  } else {                               // MODE_UP: cat(ix[r], aggr[r])
    return (k < g.D) ? g.IX + (size_t)r * g.D + k
                     : g.AGG + (size_t)r * g.Hagg + (k - g.D);
  }
}

__device__ inline void store_tile(const GemmArgs& g, const f32x8& acc, int tile,
                                  int rowbase, int m, int hi) {
  const int n = (tile << 4) + m;
  const float bv = g.bias ? g.bias[n] : 0.0f;
#pragma unroll
  for (int v = 0; v < 8; ++v) {
    int row = rowbase + v + (hi << 3);
    size_t off = (size_t)row * g.Ncols + n;
    float x = acc[v] + bv;
    if (g.epi == EPI_GELU)       x = gelu_exact(x);
    else if (g.epi == EPI_LEAKY) x = (x > 0.0f) ? x : 0.01f * x;
    else if (g.epi == EPI_ADD)   x += g.C[off];
    g.C[off] = x;
  }
}

// ---------------------------------------------------------------------------
// Weight pre-pack: W fp32 [K, Ncols] -> bf16 in B-fragment order:
//   WP[((tile*(K/32) + kb)*32 + lane)*16 + e] = bf16(W[(kb*32 + hi*16 + e)*Ncols
//                                                     + tile*16 + (lane&15)])
// so each lane's 16 B-halfs for one WMMA are 32 contiguous bytes.
// ---------------------------------------------------------------------------
__global__ void k_pack_w(const float* __restrict__ W, unsigned short* __restrict__ WP,
                         int K, int Ncols) {
  int idx = blockIdx.x * blockDim.x + threadIdx.x;
  if (idx >= K * Ncols) return;
  int e    = idx & 15;
  int lane = (idx >> 4) & 31;
  int blk  = idx >> 9;             // tile*(K/32) + kb
  int KB   = K >> 5;
  int tile = blk / KB;
  int kb   = blk - tile * KB;
  int n    = tile * 16 + (lane & 15);
  int kk   = kb * 32 + ((lane >> 4) << 4) + e;
  WP[idx] = bfbits(W[(size_t)kk * Ncols + n]);
}

// ---------------------------------------------------------------------------
// WMMA GEMM: C[M, Ncols] = epi(gatherA[M, K] @ W[K, Ncols] + bias)
// Block: 256 threads (8 wave32s) -> 64-row tile; wave owns TPW col tiles and
// 4 row sub-tiles. M % 64 == 0, K % 32 == 0, Ncols == TPW * 128.
// ---------------------------------------------------------------------------
template <int MODE, int TPW>
__global__ __launch_bounds__(256) void wmma_gemm(GemmArgs g) {
  __shared__ __align__(16) unsigned short As[64 * 32];  // bf16 bits, 64x32 tile
  const int row0 = blockIdx.x * 64;
  const int tid  = threadIdx.x;
  const int wave = tid >> 5;
  const int lane = tid & 31;
  const int m    = lane & 15;
  const int hi   = lane >> 4;
  const int KB   = g.K >> 5;
  // staging assignment: thread -> (row i, 8-wide k-run j0)
  const int si = tid >> 2;          // 0..63
  const int sj = (tid & 3) << 3;    // 0,8,16,24

  f32x8 acc0[4] = {};
  f32x8 acc1[4] = {};

  for (int kb = 0; kb < KB; ++kb) {
    const int k0 = kb << 5;
    // cooperative A-tile stage: 2 x global_load_b128 + 1 x ds_store_b128
    {
      const float* p = rowptr8<MODE>(g, row0 + si, k0 + sj);
      f32x4 v0 = *(const f32x4*)p;
      f32x4 v1 = *(const f32x4*)(p + 4);
      union { unsigned short us[8]; u32x4 q; } pk;
#pragma unroll
      for (int e = 0; e < 4; ++e) {
        pk.us[e]     = bfbits(v0[e]);
        pk.us[4 + e] = bfbits(v1[e]);
      }
      *(u32x4*)&As[si * 32 + sj] = pk.q;
    }
    __syncthreads();

    // B fragments: 32 contiguous bytes per lane -> two global_load_b128
    BFVec b0, b1;
    {
      const u32x4* bp =
          (const u32x4*)(g.WP + (((size_t)(wave * KB + kb) * 32 + lane) << 4));
      b0.q[0] = bp[0];
      b0.q[1] = bp[1];
    }
    if (TPW == 2) {
      const u32x4* bp =
          (const u32x4*)(g.WP + (((size_t)((wave + 8) * KB + kb) * 32 + lane) << 4));
      b1.q[0] = bp[0];
      b1.q[1] = bp[1];
    }
    if (kb + 1 < KB)
      __builtin_prefetch(
          (const void*)(g.WP + (((size_t)(wave * KB + kb + 1) * 32 + lane) << 4)),
          0, 1);

    // A fragment = two aligned 16B runs of the row-major LDS tile:
    //   us[0..7]  = As[row*32 + hi*16 + 0..7]     (k = hi*16 + e)
    //   us[8..15] = As[row*32 + hi*16 + 8 + 0..7] (k = hi*16 + 8 + e)
#pragma unroll
    for (int mt = 0; mt < 4; ++mt) {
      const int row = (mt * 16 + m) * 32;
      BFVec a;
      a.q[0] = *(const u32x4*)&As[row + hi * 16];
      a.q[1] = *(const u32x4*)&As[row + hi * 16 + 8];
      acc0[mt] = __builtin_amdgcn_wmma_f32_16x16x32_bf16(
          false, a.v, false, b0.v, (short)0, acc0[mt], false, false);
      if (TPW == 2)
        acc1[mt] = __builtin_amdgcn_wmma_f32_16x16x32_bf16(
            false, a.v, false, b1.v, (short)0, acc1[mt], false, false);
    }
    __syncthreads();
  }

  // epilogue: fully unrolled, constant register indices -> no scratch spills
#pragma unroll
  for (int mt = 0; mt < 4; ++mt)
    store_tile(g, acc0[mt], wave, row0 + mt * 16, m, hi);
  if (TPW == 2) {
#pragma unroll
    for (int mt = 0; mt < 4; ++mt)
      store_tile(g, acc1[mt], wave + 8, row0 + mt * 16, m, hi);
  }
}

// ---------------------------------------------------------------------------
// Small supporting kernels
// ---------------------------------------------------------------------------
__global__ void k_build_ix(const float* __restrict__ X, const float* __restrict__ pos,
                           float* __restrict__ IX, int n_rows, int F, int P) {
  int D = F + P;
  int t = blockIdx.x * blockDim.x + threadIdx.x;
  if (t >= n_rows * D) return;
  int n = t / D, j = t - n * D;
  IX[t] = (j < F) ? X[(size_t)n * F + j] : pos[(n % 9) * P + (j - F)];
}

__global__ void k_logits(const float* __restrict__ A, int Hc,
                         const float* __restrict__ W2, const float* __restrict__ b2,
                         float* __restrict__ logit, int E) {
  int e = blockIdx.x * blockDim.x + threadIdx.x;
  if (e >= E) return;
  const f32x4* row = (const f32x4*)(A + (size_t)e * Hc);
  const f32x4* w   = (const f32x4*)W2;
  float acc = 0.0f;
#pragma unroll 4
  for (int h = 0; h < (Hc >> 2); ++h) {
    f32x4 r = row[h], ww = w[h];
    acc = fmaf(r[0], ww[0], acc);
    acc = fmaf(r[1], ww[1], acc);
    acc = fmaf(r[2], ww[2], acc);
    acc = fmaf(r[3], ww[3], acc);
  }
  logit[e] = acc + b2[0];
}

// edge id for (source s -> target d) within graph g: structural, no CSR needed
__device__ inline int edge_id(int gph, int s, int d) {
  return gph * 72 + s * 8 + (d < s ? d : d - 1);
}

__global__ void k_node_softmax(const float* __restrict__ logit,
                               float* __restrict__ wmax, float* __restrict__ wsum,
                               int n_nodes) {
  int n = blockIdx.x * blockDim.x + threadIdx.x;
  if (n >= n_nodes) return;
  int gph = n / 9, d = n - gph * 9;
  float mx = -3.0e38f;
#pragma unroll
  for (int s = 0; s < 9; ++s) {
    if (s == d) continue;
    mx = fmaxf(mx, logit[edge_id(gph, s, d)]);
  }
  float sum = 0.0f;
#pragma unroll
  for (int s = 0; s < 9; ++s) {
    if (s == d) continue;
    sum += expf(logit[edge_id(gph, s, d)] - mx);
  }
  wmax[n] = mx;
  wsum[n] = sum;
}

__global__ void k_aggr(const float* __restrict__ logit, const float* __restrict__ wmax,
                       const float* __restrict__ wsum, const float* __restrict__ rel,
                       int Hc, float* __restrict__ aggr, int n_nodes) {
  int t = blockIdx.x * blockDim.x + threadIdx.x;
  if (t >= n_nodes * Hc) return;
  int n = t / Hc, h = t - n * Hc;
  int gph = n / 9, d = n - gph * 9;
  float mx = wmax[n], inv = 1.0f / wsum[n];
  float acc = 0.0f;
#pragma unroll
  for (int s = 0; s < 9; ++s) {
    if (s == d) continue;
    int eid = edge_id(gph, s, d);
    float w = expf(logit[eid] - mx) * inv;
    acc = fmaf(w, rel[(size_t)eid * Hc + h], acc);
  }
  aggr[t] = acc;
}

// one wave32 per row, 8 rows per 256-thread block; in-place capable
__global__ __launch_bounds__(256) void k_layernorm(
    const float* __restrict__ Yv, int Hc, const float* __restrict__ gam,
    const float* __restrict__ bet, float* __restrict__ out, int ldo, int col0,
    int n_rows) {
  int wave = threadIdx.x >> 5, lane = threadIdx.x & 31;
  int row = blockIdx.x * 8 + wave;
  if (row >= n_rows) return;
  const int per = Hc >> 5;  // 4 or 8
  float vals[8];
  const float* r = Yv + (size_t)row * Hc;
  for (int i = 0; i < per; ++i) vals[i] = r[i * 32 + lane];
  float s = 0.0f;
  for (int i = 0; i < per; ++i) s += vals[i];
  for (int o = 16; o; o >>= 1) s += __shfl_xor(s, o, 32);
  float mu = s / (float)Hc;
  float v2 = 0.0f;
  for (int i = 0; i < per; ++i) { float dd = vals[i] - mu; v2 += dd * dd; }
  for (int o = 16; o; o >>= 1) v2 += __shfl_xor(v2, o, 32);
  float rs = rsqrtf(v2 / (float)Hc + 1e-5f);
  for (int i = 0; i < per; ++i) {
    int j = i * 32 + lane;
    out[(size_t)row * ldo + col0 + j] = (vals[i] - mu) * rs * gam[j] + bet[j];
  }
}

__global__ void k_pool(const float* __restrict__ h2, float* __restrict__ pool,
                       int Gn, int Hc) {
  int t = blockIdx.x * blockDim.x + threadIdx.x;
  if (t >= Gn * Hc) return;
  int g = t / Hc, h = t - g * Hc;
  float acc = 0.0f;
#pragma unroll
  for (int i = 0; i < 9; ++i) acc += h2[(size_t)(g * 9 + i) * Hc + h];
  pool[t] = acc;
}

__global__ void k_final(const float* __restrict__ cm, const float* __restrict__ W2,
                        const float* __restrict__ b2, float* __restrict__ out,
                        int Gn, int Hc) {
  int g = blockIdx.x * blockDim.x + threadIdx.x;
  if (g >= Gn) return;
  const f32x4* row = (const f32x4*)(cm + (size_t)g * Hc);
  const f32x4* w   = (const f32x4*)W2;
  float acc = 0.0f;
#pragma unroll 4
  for (int h = 0; h < (Hc >> 2); ++h) {
    f32x4 r = row[h], ww = w[h];
    acc = fmaf(r[0], ww[0], acc);
    acc = fmaf(r[1], ww[1], acc);
    acc = fmaf(r[2], ww[2], acc);
    acc = fmaf(r[3], ww[3], acc);
  }
  out[g] = acc + b2[0];
}

// ---------------------------------------------------------------------------
// Host orchestration
// ---------------------------------------------------------------------------
static inline int cdiv(int a, int b) { return (a + b - 1) / b; }

extern "C" void kernel_launch(void* const* d_in, const int* in_sizes, int n_in,
                              void* d_out, int out_size, void* d_ws, size_t ws_size,
                              hipStream_t stream) {
  (void)in_sizes; (void)n_in; (void)out_size; (void)ws_size;
  const int Gn = 2048, Nn = Gn * 9, E = Nn * 8;  // 18432 nodes, 147456 edges

  const float* f4 = (const float*)d_in[0];
  const float* f2 = (const float*)d_in[1];
  const float* f1 = (const float*)d_in[2];
  const int* ei   = (const int*)d_in[4];
  const int* src  = ei;
  const int* dst  = ei + E;
  const float* pos = (const float*)d_in[11];

  auto P = [&](int i) { return (const float*)d_in[i]; };

  // workspace layout (fp32 activations + bf16 packed-weight arena)
  float* ws = (float*)d_ws;
  float* X    = ws;                        // [N,128]
  float* IX   = X    + (size_t)Nn * 128;   // [N,448] (D max)
  float* U    = IX   + (size_t)Nn * 448;   // [N,384]
  float* EA   = U    + (size_t)Nn * 384;   // [E,256]
  float* EB   = EA   + (size_t)E  * 256;   // [E,256]
  float* LOG  = EB   + (size_t)E  * 256;   // [E]
  float* WMX  = LOG  + (size_t)E;          // [N]
  float* WSM  = WMX  + (size_t)Nn;         // [N]
  float* AGR  = WSM  + (size_t)Nn;         // [N,256]
  float* Y    = AGR  + (size_t)Nn * 256;   // [N,256]
  float* CMID = Y    + (size_t)Nn * 256;   // [G,128]
  float* POOL = CMID + (size_t)Gn * 128;   // [G,256]
  unsigned short* arena = (unsigned short*)(POOL + (size_t)Gn * 256);

  // pack all WMMA weights to bf16 B-fragment layout once per launch
  size_t arena_off = 0;
  auto pack = [&](int pidx, int K, int Nc) {
    unsigned short* wp = arena + arena_off;
    arena_off += (size_t)K * Nc;
    k_pack_w<<<cdiv(K * Nc, 256), 256, 0, stream>>>(P(pidx), wp, K, Nc);
    return (const unsigned short*)wp;
  };
  const unsigned short* WPproj[3] = {pack(5, 4096, 128), pack(7, 1024, 128),
                                     pack(9, 256, 128)};
  const unsigned short* WPg1_pw1 = pack(12, 384, 128);
  const unsigned short* WPg1_pw2 = pack(14, 128, 128);
  const unsigned short* WPg1_at1 = pack(16, 384, 128);
  const unsigned short* WPg1_up  = pack(20, 320, 128);
  const unsigned short* WPg1_sk  = pack(22, 192, 128);
  const unsigned short* WPg2_pw1 = pack(26, 896, 256);
  const unsigned short* WPg2_pw2 = pack(28, 256, 256);
  const unsigned short* WPg2_at1 = pack(30, 896, 256);
  const unsigned short* WPg2_up  = pack(34, 704, 256);
  const unsigned short* WPg2_sk  = pack(36, 448, 256);
  const unsigned short* WPcls1   = pack(40, 256, 128);

  auto gemm = [&](const float* A, const unsigned short* WP, const float* bias,
                  float* C, int M, int K, int Nc, int mode, int epi,
                  const float* IXp, int D, int F, int Pp2,
                  const float* AGG, int Hagg) {
    GemmArgs ga;
    ga.A = A; ga.WP = WP; ga.bias = bias; ga.C = C;
    ga.M = M; ga.K = K; ga.Ncols = Nc; ga.epi = epi;
    ga.src = src; ga.dst = dst;
    ga.IX = IXp; ga.D = D; ga.F = F; ga.P = Pp2;
    ga.AGG = AGG; ga.Hagg = Hagg;
    dim3 grid(M / 64), blk(256);
    if (Nc == 128) {
      switch (mode) {
        case MODE_DIRECT:  wmma_gemm<MODE_DIRECT, 1><<<grid, blk, 0, stream>>>(ga); break;
        case MODE_EDGE_PW: wmma_gemm<MODE_EDGE_PW, 1><<<grid, blk, 0, stream>>>(ga); break;
        case MODE_EDGE_AT: wmma_gemm<MODE_EDGE_AT, 1><<<grid, blk, 0, stream>>>(ga); break;
        default:           wmma_gemm<MODE_UP, 1><<<grid, blk, 0, stream>>>(ga); break;
      }
    } else {
      switch (mode) {
        case MODE_DIRECT:  wmma_gemm<MODE_DIRECT, 2><<<grid, blk, 0, stream>>>(ga); break;
        case MODE_EDGE_PW: wmma_gemm<MODE_EDGE_PW, 2><<<grid, blk, 0, stream>>>(ga); break;
        case MODE_EDGE_AT: wmma_gemm<MODE_EDGE_AT, 2><<<grid, blk, 0, stream>>>(ga); break;
        default:           wmma_gemm<MODE_UP, 2><<<grid, blk, 0, stream>>>(ga); break;
      }
    }
  };

  // ---- stage 1: shared GNN over three scales ----
  const float* feats[3] = {f4, f2, f1};
  const int    Ks[3]    = {4096, 1024, 256};
  for (int s = 0; s < 3; ++s) {
    const int gb = 12;  // g1 params base
    gemm(feats[s], WPproj[s], P(s == 0 ? 6 : (s == 1 ? 8 : 10)), X, Nn, Ks[s], 128,
         MODE_DIRECT, EPI_NONE, nullptr, 0, 0, 0, nullptr, 0);
    k_build_ix<<<cdiv(Nn * 192, 256), 256, 0, stream>>>(X, pos, IX, Nn, 128, 64);
    gemm(nullptr, WPg1_pw1, P(gb + 1), EA, E, 384, 128,
         MODE_EDGE_PW, EPI_GELU, IX, 192, 128, 64, nullptr, 0);
    gemm(EA, WPg1_pw2, P(gb + 3), EB, E, 128, 128,
         MODE_DIRECT, EPI_NONE, nullptr, 0, 0, 0, nullptr, 0);
    gemm(nullptr, WPg1_at1, P(gb + 5), EA, E, 384, 128,
         MODE_EDGE_AT, EPI_LEAKY, IX, 192, 128, 64, nullptr, 0);
    k_logits<<<cdiv(E, 256), 256, 0, stream>>>(EA, 128, P(gb + 6), P(gb + 7), LOG, E);
    k_node_softmax<<<cdiv(Nn, 256), 256, 0, stream>>>(LOG, WMX, WSM, Nn);
    k_aggr<<<cdiv(Nn * 128, 256), 256, 0, stream>>>(LOG, WMX, WSM, EB, 128, AGR, Nn);
    gemm(nullptr, WPg1_up, P(gb + 9), Y, Nn, 320, 128,
         MODE_UP, EPI_GELU, IX, 192, 0, 0, AGR, 128);
    gemm(IX, WPg1_sk, P(gb + 11), Y, Nn, 192, 128,
         MODE_DIRECT, EPI_ADD, nullptr, 0, 0, 0, nullptr, 0);
    k_layernorm<<<cdiv(Nn, 8), 256, 0, stream>>>(Y, 128, P(gb + 12), P(gb + 13),
                                                 U, 384, s * 128, Nn);
  }

  // ---- stage 2: GNN on concatenated features (F=384, Hout=256) ----
  {
    const int gb = 26;  // g2 params base
    k_build_ix<<<cdiv(Nn * 448, 256), 256, 0, stream>>>(U, pos, IX, Nn, 384, 64);
    gemm(nullptr, WPg2_pw1, P(gb + 1), EA, E, 896, 256,
         MODE_EDGE_PW, EPI_GELU, IX, 448, 384, 64, nullptr, 0);
    gemm(EA, WPg2_pw2, P(gb + 3), EB, E, 256, 256,
         MODE_DIRECT, EPI_NONE, nullptr, 0, 0, 0, nullptr, 0);
    gemm(nullptr, WPg2_at1, P(gb + 5), EA, E, 896, 256,
         MODE_EDGE_AT, EPI_LEAKY, IX, 448, 384, 64, nullptr, 0);
    k_logits<<<cdiv(E, 256), 256, 0, stream>>>(EA, 256, P(gb + 6), P(gb + 7), LOG, E);
    k_node_softmax<<<cdiv(Nn, 256), 256, 0, stream>>>(LOG, WMX, WSM, Nn);
    k_aggr<<<cdiv(Nn * 256, 256), 256, 0, stream>>>(LOG, WMX, WSM, EB, 256, AGR, Nn);
    gemm(nullptr, WPg2_up, P(gb + 9), Y, Nn, 704, 256,
         MODE_UP, EPI_GELU, IX, 448, 0, 0, AGR, 256);
    gemm(IX, WPg2_sk, P(gb + 11), Y, Nn, 448, 256,
         MODE_DIRECT, EPI_ADD, nullptr, 0, 0, 0, nullptr, 0);
    k_layernorm<<<cdiv(Nn, 8), 256, 0, stream>>>(Y, 256, P(gb + 12), P(gb + 13),
                                                 Y, 256, 0, Nn);  // in place
  }

  // ---- pool + classifier ----
  k_pool<<<cdiv(Gn * 256, 256), 256, 0, stream>>>(Y, POOL, Gn, 256);
  gemm(POOL, WPcls1, P(41), CMID, Gn, 256, 128,
       MODE_DIRECT, EPI_GELU, nullptr, 0, 0, 0, nullptr, 0);
  k_final<<<cdiv(Gn, 256), 256, 0, stream>>>(CMID, P(42), P(43), (float*)d_out, Gn, 128);
}
